// AttentionBlock_89343909691648
// MI455X (gfx1250) — matrix operand order repaired and checked by reading, not compiled, not run on hardware
//
#include <hip/hip_runtime.h>

// MI455X (gfx1250) implementation.
// bf16 WMMA (16x16x32, f32 accum) for all GEMMs; 16-row score slab held fully
// in LDS (256KB of the 320KB/WGP) so softmax never touches HBM; async
// global->LDS staging (ASYNCcnt) for tile loads. Requires ws_size >= 66 MiB.

typedef __attribute__((ext_vector_type(16))) __bf16 v16bf;
typedef __attribute__((ext_vector_type(8)))  float  v8f;

union FragBF { v16bf v; unsigned int u[8]; };
union FragF  { v8f   v; float        f[8]; };

__device__ __forceinline__ unsigned short f2bf(float f) {
  unsigned int u = __float_as_uint(f);
  u += 0x7FFFu + ((u >> 16) & 1u);          // round-to-nearest-even
  return (unsigned short)(u >> 16);
}

// gfx1250 async global->LDS copy (tracked by ASYNCcnt).
__device__ __forceinline__ void async_b128(void* lds, const void* g) {
  unsigned int l = (unsigned int)(unsigned long long)lds;  // low 32b of LDS flat addr == LDS offset
  unsigned long long ga = (unsigned long long)g;
  asm volatile("global_load_async_to_lds_b128 %0, %1, off" :: "v"(l), "v"(ga) : "memory");
}
__device__ __forceinline__ void wait_async() {
  asm volatile("s_wait_asynccnt 0x0" ::: "memory");
}

// ---------------------------------------------------------------------------
// GroupNorm statistics: one block per (b,g); 16 ch * 4096 px contiguous span.
// ---------------------------------------------------------------------------
__global__ __launch_bounds__(256) void gn_stats_kernel(const float* __restrict__ x,
                                                       float* __restrict__ mean,
                                                       float* __restrict__ rstd) {
  __shared__ float rs[256], rq[256];
  const float4* p = (const float4*)(x + (size_t)blockIdx.x * 65536);
  float s = 0.f, s2 = 0.f;
  for (int i = threadIdx.x; i < 16384; i += 256) {
    float4 v = p[i];
    s  += v.x + v.y + v.z + v.w;
    s2 += v.x*v.x + v.y*v.y + v.z*v.z + v.w*v.w;
  }
  rs[threadIdx.x] = s; rq[threadIdx.x] = s2;
  __syncthreads();
  for (int st = 128; st > 0; st >>= 1) {
    if ((int)threadIdx.x < st) {
      rs[threadIdx.x] += rs[threadIdx.x + st];
      rq[threadIdx.x] += rq[threadIdx.x + st];
    }
    __syncthreads();
  }
  if (threadIdx.x == 0) {
    float m = rs[0] * (1.f / 65536.f);
    float v = rq[0] * (1.f / 65536.f) - m * m;
    mean[blockIdx.x] = m;
    rstd[blockIdx.x] = rsqrtf(v + 1e-6f);
  }
}

// ---------------------------------------------------------------------------
// Apply GN and transpose to HT[b][p][c] in bf16 (c contiguous -> GEMM A layout)
// ---------------------------------------------------------------------------
__global__ void gn_apply_t_kernel(const float* __restrict__ x,
                                  const float* __restrict__ gamma,
                                  const float* __restrict__ beta,
                                  const float* __restrict__ mean,
                                  const float* __restrict__ rstd,
                                  unsigned short* __restrict__ HT) {
  __shared__ float tile[32][33];
  const int b = blockIdx.z, c0 = blockIdx.y * 32, p0 = blockIdx.x * 32;
  const int tx = threadIdx.x, ty = threadIdx.y;
#pragma unroll
  for (int r = 0; r < 4; ++r) {
    int cl = ty + r * 8;
    int c = c0 + cl;
    int g = c >> 4;
    float m = mean[b * 32 + g], rv = rstd[b * 32 + g];
    float v = x[((size_t)b * 512 + c) * 4096 + p0 + tx];
    tile[cl][tx] = (v - m) * rv * gamma[c] + beta[c];
  }
  __syncthreads();
#pragma unroll
  for (int r = 0; r < 4; ++r) {
    int pl = ty + r * 8;
    HT[((size_t)b * 4096 + p0 + pl) * 512 + c0 + tx] = f2bf(tile[tx][pl]);
  }
}

// ---------------------------------------------------------------------------
// Transpose 512x512 f32 weights -> bf16 WT[c][o] (o contiguous = WMMA-B layout)
// ---------------------------------------------------------------------------
__global__ void wt_transpose_kernel(const float* __restrict__ w0, const float* __restrict__ w1,
                                    const float* __restrict__ w2,
                                    unsigned short* __restrict__ o0, unsigned short* __restrict__ o1,
                                    unsigned short* __restrict__ o2) {
  const float* w = (blockIdx.z == 0) ? w0 : (blockIdx.z == 1) ? w1 : w2;
  unsigned short* o = (blockIdx.z == 0) ? o0 : (blockIdx.z == 1) ? o1 : o2;
  __shared__ float tile[32][33];
  const int r0 = blockIdx.y * 32, c0 = blockIdx.x * 32;
  const int tx = threadIdx.x, ty = threadIdx.y;
#pragma unroll
  for (int r = 0; r < 4; ++r) {
    int rl = ty + r * 8;
    tile[rl][tx] = w[(size_t)(r0 + rl) * 512 + c0 + tx];
  }
  __syncthreads();
#pragma unroll
  for (int r = 0; r < 4; ++r) {
    int cl = ty + r * 8;
    o[(size_t)(c0 + cl) * 512 + r0 + tx] = f2bf(tile[tx][cl]);
  }
}

// ---------------------------------------------------------------------------
// C[p,n] = A[b][4096x512] x B[512x512] (+bias). mode 0: bf16 out [b][p][n].
// mode 1: f32 out[b][n][p] = acc + bias[n] + resid[b][n][p]  (final conv).
// 8 waves: wave w owns M-tile w (16 rows) x 64 cols (4 WMMA N-tiles).
// ---------------------------------------------------------------------------
__global__ __launch_bounds__(256) void gemm_nn_bf16(
    const unsigned short* __restrict__ A,
    const unsigned short* __restrict__ B,
    const float* __restrict__ bias,
    unsigned short* __restrict__ Cbf,
    float* __restrict__ Cf32,
    const float* __restrict__ resid,
    int mode) {
  __shared__ unsigned short Ab[128 * 32];
  __shared__ unsigned short Bb[32 * 64];
  const int t = threadIdx.x, wave = t >> 5, lane = t & 31;
  const int b = blockIdx.z, m0 = blockIdx.x * 128, n0 = blockIdx.y * 64;
  const unsigned short* Abase = A + ((size_t)b * 4096 + m0) * 512;
  const int kb = (lane < 16) ? 0 : 8;
  const int mrow = lane & 15;
  const int mofs = (lane < 16) ? 0 : 8;

  FragF acc[4];
#pragma unroll
  for (int i = 0; i < 4; ++i)
#pragma unroll
    for (int r = 0; r < 8; ++r) acc[i].f[r] = 0.f;

  for (int kk = 0; kk < 16; ++kk) {
    // stage A block [128 x 32] via async copy, B block [32 x 64] via plain copy
    {
      int s0 = t * 2;
#pragma unroll
      for (int q = 0; q < 2; ++q) {
        int s = s0 + q, row = s >> 2, seg = s & 3;
        async_b128(Ab + row * 32 + seg * 8, Abase + (size_t)row * 512 + kk * 32 + seg * 8);
      }
      int row = t >> 3, seg = t & 7;
      *(uint4*)(Bb + row * 64 + seg * 8) =
          *(const uint4*)(B + (size_t)(kk * 32 + row) * 512 + n0 + seg * 8);
    }
    wait_async();
    __syncthreads();

    FragBF af;
    const unsigned int* ap = (const unsigned int*)(Ab + (wave * 16 + mrow) * 32);
#pragma unroll
    for (int q = 0; q < 4; ++q) { af.u[q] = ap[(kb >> 1) + q]; af.u[4 + q] = ap[(kb >> 1) + 8 + q]; }
#pragma unroll
    for (int nt = 0; nt < 4; ++nt) {
      FragBF bfm;
      const unsigned short* bp = Bb + nt * 16 + mrow;
#pragma unroll
      for (int q = 0; q < 8; ++q) {
        int k = (q < 4) ? kb + 2 * q : 16 + kb + 2 * (q - 4);
        bfm.u[q] = (unsigned int)bp[k * 64] | ((unsigned int)bp[(k + 1) * 64] << 16);
      }
      acc[nt].v = __builtin_amdgcn_wmma_f32_16x16x32_bf16(false, af.v, false, bfm.v,
                                                          (short)0, acc[nt].v, false, false);
    }
    __syncthreads();
  }

  const int mbase = m0 + wave * 16 + mofs;
#pragma unroll
  for (int nt = 0; nt < 4; ++nt) {
    int col = n0 + nt * 16 + mrow;
    float bvv = bias[col];
    if (mode == 0) {
#pragma unroll
      for (int r = 0; r < 8; ++r)
        Cbf[((size_t)b * 4096 + mbase + r) * 512 + col] = f2bf(acc[nt].f[r] + bvv);
    } else {
#pragma unroll
      for (int r = 0; r < 8; ++r) {
        size_t oi = ((size_t)b * 512 + col) * 4096 + (size_t)(mbase + r);
        Cf32[oi] = acc[nt].f[r] + bvv + resid[oi];
      }
    }
  }
}

// ---------------------------------------------------------------------------
// Fused attention: one WG per 16-row block. S slab [16 x 4096] f32 lives in
// LDS; K2 is the QT buffer reinterpreted as [512 x 4096] (reference reshape
// bug). Softmax (scale -256) in LDS, then O = P x VT via WMMA.
// ---------------------------------------------------------------------------
__global__ __launch_bounds__(256) void attn_kernel(const unsigned short* __restrict__ QT,
                                                   const unsigned short* __restrict__ VT,
                                                   unsigned short* __restrict__ OT) {
  extern __shared__ char smem[];
  float* S            = (float*)smem;                         // 16*4096*4 = 262144
  unsigned short* Qb  = (unsigned short*)(smem + 262144);     // 16*512*2  = 16384
  unsigned short* Pb  = (unsigned short*)(smem + 278528);     // 16*512*2  = 16384
  float* red          = (float*)(smem + 294912);              // 16*16*4   = 1024
  float* rowm         = (float*)(smem + 295936);              // 64
  float* rowl         = (float*)(smem + 296000);              // 64

  const int t = threadIdx.x;
  const int wave = t >> 5, lane = t & 31;
  const int b = blockIdx.y, rb = blockIdx.x;
  const unsigned short* qb = QT + (size_t)b * (4096 * 512);
  const unsigned short* vb = VT + (size_t)b * (4096 * 512);
  const int kb = (lane < 16) ? 0 : 8;
  const int mrow = lane & 15;
  const int mofs = (lane < 16) ? 0 : 8;

  // ---- load Q row-block [16 x 512] into LDS (async) ----
#pragma unroll
  for (int q = 0; q < 4; ++q) {
    int s = t * 4 + q;
    int row = s >> 6, off = (s & 63) * 8;
    async_b128(Qb + row * 512 + off, qb + (size_t)(rb * 16 + row) * 512 + off);
  }
  wait_async();
  __syncthreads();

  // ---- phase 1: S = -256 * (Q x K2); each wave owns a 512-col slab ----
  FragBF af[16];
  {
    const unsigned int* ap = (const unsigned int*)Qb + mrow * 256;
#pragma unroll
    for (int kk = 0; kk < 16; ++kk) {
      const unsigned int* p = ap + kk * 16;
#pragma unroll
      for (int q = 0; q < 4; ++q) { af[kk].u[q] = p[(kb >> 1) + q]; af[kk].u[4 + q] = p[(kb >> 1) + 8 + q]; }
    }
  }
  const int j0w = wave * 512;
  for (int nt = 0; nt < 32; ++nt) {
    int j = j0w + nt * 16;
    FragF acc;
#pragma unroll
    for (int r = 0; r < 8; ++r) acc.f[r] = 0.f;
#pragma unroll
    for (int kk = 0; kk < 16; ++kk) {
      FragBF bfm;  // K2[c, j] == QT_flat[c*4096 + j]
      const unsigned short* bp = qb + (size_t)(kk * 32) * 4096 + j + mrow;
#pragma unroll
      for (int q = 0; q < 8; ++q) {
        int k = (q < 4) ? kb + 2 * q : 16 + kb + 2 * (q - 4);
        bfm.u[q] = (unsigned int)bp[(size_t)k * 4096] | ((unsigned int)bp[(size_t)(k + 1) * 4096] << 16);
      }
      acc.v = __builtin_amdgcn_wmma_f32_16x16x32_bf16(false, af[kk].v, false, bfm.v,
                                                      (short)0, acc.v, false, false);
    }
    int n = j + mrow;
#pragma unroll
    for (int r = 0; r < 8; ++r) S[(size_t)(r + mofs) * 4096 + n] = acc.f[r] * -256.0f;
  }
  __syncthreads();

  // ---- phase 2: softmax stats over LDS; store exp(s-m) back in place ----
  {
    int row = t >> 4, w16 = t & 15;
    float* srow = S + (size_t)row * 4096;
    float m = -1e30f;
    for (int j = w16; j < 4096; j += 16) m = fmaxf(m, srow[j]);
    red[row * 16 + w16] = m;
    __syncthreads();
    if (w16 == 0) {
      float mm = -1e30f;
      for (int i = 0; i < 16; ++i) mm = fmaxf(mm, red[row * 16 + i]);
      rowm[row] = mm;
    }
    __syncthreads();
    float mm = rowm[row];
    float sum = 0.f;
    for (int j = w16; j < 4096; j += 16) {
      float e = __expf(srow[j] - mm);
      srow[j] = e;
      sum += e;
    }
    red[row * 16 + w16] = sum;
    __syncthreads();
    if (w16 == 0) {
      float ss = 0.f;
      for (int i = 0; i < 16; ++i) ss += red[row * 16 + i];
      rowl[row] = 1.f / ss;
    }
    __syncthreads();
  }

  // ---- phase 3: O = P x VT; wave owns 64-col output slab; j in 512-chunks --
  FragF oacc[4];
#pragma unroll
  for (int i = 0; i < 4; ++i)
#pragma unroll
    for (int r = 0; r < 8; ++r) oacc[i].f[r] = 0.f;
  const int c0 = wave * 64;
  for (int cj = 0; cj < 8; ++cj) {
    for (int idx = t; idx < 8192; idx += 256) {
      int rr = idx >> 9, jj = idx & 511;
      Pb[rr * 512 + jj] = f2bf(S[(size_t)rr * 4096 + cj * 512 + jj] * rowl[rr]);
    }
    __syncthreads();
    for (int kk = 0; kk < 16; ++kk) {
      FragBF pa;
      const unsigned int* p = (const unsigned int*)Pb + mrow * 256 + kk * 16;
#pragma unroll
      for (int q = 0; q < 4; ++q) { pa.u[q] = p[(kb >> 1) + q]; pa.u[4 + q] = p[(kb >> 1) + 8 + q]; }
      int jg = cj * 512 + kk * 32;
      const unsigned short* vrow = vb + (size_t)jg * 512 + c0 + mrow;
#pragma unroll
      for (int nt = 0; nt < 4; ++nt) {
        FragBF bfm;
        const unsigned short* bp = vrow + nt * 16;
#pragma unroll
        for (int q = 0; q < 8; ++q) {
          int k = (q < 4) ? kb + 2 * q : 16 + kb + 2 * (q - 4);
          bfm.u[q] = (unsigned int)bp[(size_t)k * 512] | ((unsigned int)bp[(size_t)(k + 1) * 512] << 16);
        }
        oacc[nt].v = __builtin_amdgcn_wmma_f32_16x16x32_bf16(false, pa.v, false, bfm.v,
                                                             (short)0, oacc[nt].v, false, false);
      }
    }
    __syncthreads();
  }

  // ---- epilogue: store O^T tile (bf16, [p][c] layout for the final GEMM) ----
  size_t obase = ((size_t)b * 4096 + rb * 16) * 512;
#pragma unroll
  for (int nt = 0; nt < 4; ++nt) {
    int col = c0 + nt * 16 + mrow;
#pragma unroll
    for (int r = 0; r < 8; ++r)
      OT[obase + (size_t)(r + mofs) * 512 + col] = f2bf(oacc[nt].f[r]);
  }
}

// ---------------------------------------------------------------------------
extern "C" void kernel_launch(void* const* d_in, const int* in_sizes, int n_in,
                              void* d_out, int out_size, void* d_ws, size_t ws_size,
                              hipStream_t stream) {
  (void)in_sizes; (void)n_in; (void)out_size; (void)ws_size;
  const float* x     = (const float*)d_in[0];
  const float* gamma = (const float*)d_in[1];
  const float* beta  = (const float*)d_in[2];
  const float* wq    = (const float*)d_in[3];
  const float* bq    = (const float*)d_in[4];
  // d_in[5] (wk), d_in[6] (bk) are dead code in the reference
  const float* wv    = (const float*)d_in[7];
  const float* bv    = (const float*)d_in[8];
  const float* wo    = (const float*)d_in[9];
  const float* bo    = (const float*)d_in[10];
  float* out = (float*)d_out;

  char* ws = (char*)d_ws;
  const size_t MB = 1ull << 20;
  float* mean = (float*)(ws);
  float* rstd = (float*)(ws + 512);
  unsigned short* WqT = (unsigned short*)(ws + 4096);
  unsigned short* WvT = (unsigned short*)(ws + 4096 + 524288);
  unsigned short* WoT = (unsigned short*)(ws + 4096 + 2 * 524288);
  unsigned short* HT  = (unsigned short*)(ws + 2 * MB);   // [4][4096][512] bf16
  unsigned short* QT  = (unsigned short*)(ws + 18 * MB);  // [4][4096][512] bf16 (also K2)
  unsigned short* VT  = (unsigned short*)(ws + 34 * MB);  // [4][4096][512] bf16
  unsigned short* OT  = (unsigned short*)(ws + 50 * MB);  // [4][4096][512] bf16
  // requires ws_size >= 66 MiB

  gn_stats_kernel<<<dim3(128), dim3(256), 0, stream>>>(x, mean, rstd);
  gn_apply_t_kernel<<<dim3(128, 16, 4), dim3(32, 8), 0, stream>>>(x, gamma, beta, mean, rstd, HT);
  wt_transpose_kernel<<<dim3(16, 16, 3), dim3(32, 8), 0, stream>>>(wq, wv, wo, WqT, WvT, WoT);
  gemm_nn_bf16<<<dim3(32, 8, 4), dim3(256), 0, stream>>>(HT, WqT, bq, QT, nullptr, nullptr, 0);
  gemm_nn_bf16<<<dim3(32, 8, 4), dim3(256), 0, stream>>>(HT, WvT, bv, VT, nullptr, nullptr, 0);
  attn_kernel<<<dim3(256, 4), dim3(256), 296064, stream>>>(QT, VT, OT);
  gemm_nn_bf16<<<dim3(32, 8, 4), dim3(256), 0, stream>>>(OT, WoT, bo, nullptr, out, x, 1);
}
